// PairBiasSelfAttention_55405078119354
// MI455X (gfx1250) — compile-verified, matrix operand order checked by reading, not run on hardware
//
#include <hip/hip_runtime.h>
#include <hip/hip_bf16.h>
#include <stdint.h>

// Problem constants (match reference)
#define B_  4
#define N_  1024
#define C_  512
#define H_  8
#define D_  64
#define P_  64
#define BN_ (B_ * N_)          // 4096 rows in the token GEMMs

typedef unsigned short u16;
typedef __attribute__((ext_vector_type(16))) __bf16 v16bf;
typedef __attribute__((ext_vector_type(8)))  float  v8f;

struct Frag {
  union { v16bf v; uint4 q[2]; };
};

__device__ __forceinline__ u16 f2bf(float f) {
  uint32_t u = __float_as_uint(f);
  u += 0x7fffu + ((u >> 16) & 1u);   // round-to-nearest-even
  return (u16)(u >> 16);
}

// CDNA5 wave32 16-bit A/B fragment (16x32). Per ISA 7.12.2:
// lane l<16  holds row (l),    K = {0..7} in v0..3, {16..23} in v4..7
// lane l>=16 holds row (l-16), K = {8..15},          {24..31}
// => per lane: 8 contiguous bf16 at col (hi*8), 8 more at col (16 + hi*8).
__device__ __forceinline__ Frag load_frag(const u16* tile, int stride) {
  int lane = threadIdx.x & 31;
  int l  = lane & 15;
  int hi = lane >> 4;
  const u16* p = tile + (size_t)l * stride + hi * 8;
  Frag f;
  f.q[0] = *(const uint4*)(p);
  f.q[1] = *(const uint4*)(p + 16);
  return f;
}

__device__ __forceinline__ v8f wmma_bf16(const Frag& a, const Frag& b, v8f c) {
  return __builtin_amdgcn_wmma_f32_16x16x32_bf16(false, a.v, false, b.v,
                                                 (short)0, c, false, false);
}

// ---------------------------------------------------------------------------
// Conversion helpers
// ---------------------------------------------------------------------------
__global__ void f32_to_bf16_kernel(const float* __restrict__ in,
                                   u16* __restrict__ out, int count) {
  int i = blockIdx.x * blockDim.x + threadIdx.x;
  if (i < count) out[i] = f2bf(in[i]);
}

// in[K][Ncols] f32 -> out[Ncols][K] bf16 (pre-transpose weights so B-fragments
// are contiguous row loads)
__global__ void transpose_to_bf16_kernel(const float* __restrict__ in,
                                         u16* __restrict__ out,
                                         int K, int Ncols) {
  int i = blockIdx.x * blockDim.x + threadIdx.x;
  if (i < K * Ncols) {
    int k = i / Ncols, n = i % Ncols;
    out[(size_t)n * K + k] = f2bf(in[i]);
  }
}

// ---------------------------------------------------------------------------
// pair_bias[h][n][m] = sum_p pair_features[n][m][p] * Wp[p][h]
// Pure streaming kernel: 256 MB read once, 32 MB written. Memory bound.
// ---------------------------------------------------------------------------
__global__ void pair_bias_kernel(const float* __restrict__ pf,
                                 const float* __restrict__ Wp,
                                 float* __restrict__ bias) {
  __shared__ float wp[P_ * H_];
  for (int i = threadIdx.x; i < P_ * H_; i += blockDim.x) wp[i] = Wp[i];
  __syncthreads();

  int idx = blockIdx.x * blockDim.x + threadIdx.x;   // idx = n*N + m
  const float* row = pf + (size_t)idx * P_;
  if (idx + 256 < N_ * N_)
    __builtin_prefetch(pf + (size_t)(idx + 256) * P_, 0, 0);  // global_prefetch

  float acc[H_];
#pragma unroll
  for (int h = 0; h < H_; ++h) acc[h] = 0.0f;

#pragma unroll
  for (int p4 = 0; p4 < P_ / 4; ++p4) {
    float4 x = ((const float4*)row)[p4];
    const float* w0 = &wp[(p4 * 4 + 0) * H_];
    const float* w1 = &wp[(p4 * 4 + 1) * H_];
    const float* w2 = &wp[(p4 * 4 + 2) * H_];
    const float* w3 = &wp[(p4 * 4 + 3) * H_];
#pragma unroll
    for (int h = 0; h < H_; ++h)
      acc[h] = fmaf(x.x, w0[h], fmaf(x.y, w1[h], fmaf(x.z, w2[h], fmaf(x.w, w3[h], acc[h]))));
  }
#pragma unroll
  for (int h = 0; h < H_; ++h)
    bias[(size_t)h * N_ * N_ + idx] = acc[h];
}

// ---------------------------------------------------------------------------
// Generic bf16 WMMA GEMM with software-pipelined (double-buffered) K loop.
// out[m][n] = A[m][:] . WT[n][:] + bias[n]
// Wave tile: 32M x 64N  (2 A-frags, 4 B-frags, 8 WMMAs per 32-K step).
// mode 0: q -> q_bf[b][h][n][d] scaled by 1/sqrt(d)
// mode 1: k -> k_bf[b][h][n][d]
// mode 2: v -> vT_bf[b][h][d][n]   (transposed for the P@V B-fragment)
// mode 3: out-proj -> d_out fp32 [b][n][c]
// ---------------------------------------------------------------------------
__global__ void gemm_qkvo_kernel(const u16* __restrict__ A,
                                 const u16* __restrict__ WT,
                                 const float* __restrict__ bias,
                                 void* __restrict__ outp,
                                 int mode, float scale) {
  int wave  = threadIdx.x >> 5;
  int mbase = blockIdx.x * 128 + wave * 32;  // 4 waves x 32 rows = 128 rows/blk
  int nbase = blockIdx.y * 64;

  const u16* Ar0 = A + (size_t)mbase * C_;
  const u16* Ar1 = A + (size_t)(mbase + 16) * C_;
  const u16* Wr0 = WT + (size_t)(nbase +  0) * C_;
  const u16* Wr1 = WT + (size_t)(nbase + 16) * C_;
  const u16* Wr2 = WT + (size_t)(nbase + 32) * C_;
  const u16* Wr3 = WT + (size_t)(nbase + 48) * C_;

  v8f acc[2][4] = {};

  // prologue: stage 0
  Frag a0 = load_frag(Ar0, C_), a1 = load_frag(Ar1, C_);
  Frag b0 = load_frag(Wr0, C_), b1 = load_frag(Wr1, C_);
  Frag b2 = load_frag(Wr2, C_), b3 = load_frag(Wr3, C_);

  for (int kk = 32; kk <= C_; kk += 32) {
    int kn = (kk < C_) ? kk : 0;   // wrap on last iter (dead loads, no UB)
    // issue next stage's loads before this stage's WMMAs
    Frag na0 = load_frag(Ar0 + kn, C_), na1 = load_frag(Ar1 + kn, C_);
    Frag nb0 = load_frag(Wr0 + kn, C_), nb1 = load_frag(Wr1 + kn, C_);
    Frag nb2 = load_frag(Wr2 + kn, C_), nb3 = load_frag(Wr3 + kn, C_);

    acc[0][0] = wmma_bf16(a0, b0, acc[0][0]);
    acc[1][0] = wmma_bf16(a1, b0, acc[1][0]);
    acc[0][1] = wmma_bf16(a0, b1, acc[0][1]);
    acc[1][1] = wmma_bf16(a1, b1, acc[1][1]);
    acc[0][2] = wmma_bf16(a0, b2, acc[0][2]);
    acc[1][2] = wmma_bf16(a1, b2, acc[1][2]);
    acc[0][3] = wmma_bf16(a0, b3, acc[0][3]);
    acc[1][3] = wmma_bf16(a1, b3, acc[1][3]);

    a0 = na0; a1 = na1;
    b0 = nb0; b1 = nb1; b2 = nb2; b3 = nb3;
  }

  int lane = threadIdx.x & 31;
  int ln = lane & 15, hi = lane >> 4;
#pragma unroll
  for (int t = 0; t < 2; ++t) {
#pragma unroll
    for (int c = 0; c < 4; ++c) {
#pragma unroll
      for (int r = 0; r < 8; ++r) {
        int m  = mbase + t * 16 + r + 8 * hi;   // global row (C-layout)
        int cg = nbase + c * 16 + ln;           // global col
        float val = acc[t][c][r] + bias[cg];
        if (mode == 3) {
          ((float*)outp)[(size_t)m * C_ + cg] = val;
        } else {
          int b = m >> 10, n = m & (N_ - 1);
          int h = cg >> 6, d = cg & (D_ - 1);
          u16 bv = f2bf(val * scale);
          if (mode == 2)
            ((u16*)outp)[(((size_t)(b * H_ + h)) * D_ + d) * N_ + n] = bv;
          else
            ((u16*)outp)[(((size_t)(b * H_ + h)) * N_ + n) * D_ + d] = bv;
        }
      }
    }
  }
}

// ---------------------------------------------------------------------------
// Flash attention with additive pair bias.
// Block: 128 threads (4 waves). Each wave owns 16 query rows of one (b,h).
// Streams keys in blocks of 32; online softmax; O kept in 4 f32 C-tiles.
// K fragments double-buffered across iterations; bias/V loads issued early so
// they overlap the WMMAs / softmax VALU chain.
// ---------------------------------------------------------------------------
__global__ void attn_kernel(const u16* __restrict__ qf,
                            const u16* __restrict__ kf,
                            const u16* __restrict__ vtf,
                            const float* __restrict__ bias,
                            u16* __restrict__ merged) {
  int w     = threadIdx.x >> 5;
  int qbase = blockIdx.x * 64 + w * 16;
  int hh    = blockIdx.y;
  int bb    = blockIdx.z;

  const u16*   qp    = qf  + (size_t)(bb * H_ + hh) * N_ * D_;
  const u16*   kp    = kf  + (size_t)(bb * H_ + hh) * N_ * D_;
  const u16*   vt    = vtf + (size_t)(bb * H_ + hh) * D_ * N_;
  const float* biasp = bias + (size_t)hh * N_ * N_;

  int lane = threadIdx.x & 31;
  int ln = lane & 15, hi = lane >> 4;

  // Q fragments: rows qbase..qbase+15, K-dim chunks d=[0,32) and [32,64)
  Frag qa[2];
  qa[0] = load_frag(qp + (size_t)qbase * D_ + 0,  D_);
  qa[1] = load_frag(qp + (size_t)qbase * D_ + 32, D_);

  v8f o[4] = {};
  float rmax[8], rsum[8];
#pragma unroll
  for (int r = 0; r < 8; ++r) { rmax[r] = -3.0e38f; rsum[r] = 0.0f; }

  __shared__ u16 pls[4][16 * 32];   // wave-private P-transpose tiles
  u16* myp = pls[w];

  // prologue: K fragments for jj = 0
  Frag kb[2][2];
#pragma unroll
  for (int half = 0; half < 2; ++half)
#pragma unroll
    for (int dc = 0; dc < 2; ++dc)
      kb[half][dc] = load_frag(kp + (size_t)(half * 16) * D_ + dc * 32, D_);

  for (int jj = 0; jj < N_; jj += 32) {
    // --- issue bias loads first: latency overlaps the S WMMAs ---
    float bv[2][8];
#pragma unroll
    for (int half = 0; half < 2; ++half)
#pragma unroll
      for (int r = 0; r < 8; ++r)
        bv[half][r] = biasp[(size_t)(qbase + r + 8 * hi) * N_ + jj + half * 16 + ln];

    // --- S = Q @ K^T (two 16x16 tiles for the 32 keys) ---
    v8f s[2] = {};
#pragma unroll
    for (int half = 0; half < 2; ++half)
#pragma unroll
      for (int dc = 0; dc < 2; ++dc)
        s[half] = wmma_bf16(qa[dc], kb[half][dc], s[half]);

    // --- prefetch next K block + this block's V: overlap softmax chain ---
    int jn = (jj + 32 < N_) ? jj + 32 : 0;
    Frag nk[2][2];
#pragma unroll
    for (int half = 0; half < 2; ++half)
#pragma unroll
      for (int dc = 0; dc < 2; ++dc)
        nk[half][dc] = load_frag(kp + (size_t)(jn + half * 16) * D_ + dc * 32, D_);
    Frag vb[4];
#pragma unroll
    for (int c = 0; c < 4; ++c)
      vb[c] = load_frag(vt + (size_t)(c * 16) * N_ + jj, N_);

    // --- online softmax (rows live on 16-lane groups; width-16 reductions) ---
#pragma unroll
    for (int r = 0; r < 8; ++r) {
      float s0 = s[0][r] + bv[0][r];
      float s1 = s[1][r] + bv[1][r];
      float m2 = fmaxf(s0, s1);
#pragma unroll
      for (int off = 1; off < 16; off <<= 1)
        m2 = fmaxf(m2, __shfl_xor(m2, off, 16));
      float mnew  = fmaxf(rmax[r], m2);
      float alpha = __expf(rmax[r] - mnew);
      rmax[r] = mnew;
      float p0 = __expf(s0 - mnew);
      float p1 = __expf(s1 - mnew);
      s[0][r] = p0; s[1][r] = p1;
      float ps = p0 + p1;
#pragma unroll
      for (int off = 1; off < 16; off <<= 1)
        ps += __shfl_xor(ps, off, 16);
      rsum[r] = rsum[r] * alpha + ps;
#pragma unroll
      for (int c = 0; c < 4; ++c) o[c][r] *= alpha;
    }

    // --- re-layout P: C-layout -> A-fragment via wave-private LDS tile ---
#pragma unroll
    for (int half = 0; half < 2; ++half)
#pragma unroll
      for (int r = 0; r < 8; ++r)
        myp[(r + 8 * hi) * 32 + half * 16 + ln] = f2bf(s[half][r]);
    __builtin_amdgcn_wave_barrier();
    Frag pa;
    {
      const u16* p = &myp[(size_t)ln * 32 + hi * 8];
      pa.q[0] = *(const uint4*)(p);
      pa.q[1] = *(const uint4*)(p + 16);
    }
    __builtin_amdgcn_wave_barrier();

    // --- O += P @ V  (Vt[d][n] rows give contiguous B-fragments) ---
#pragma unroll
    for (int c = 0; c < 4; ++c)
      o[c] = wmma_bf16(pa, vb[c], o[c]);

    // rotate K double buffer
#pragma unroll
    for (int half = 0; half < 2; ++half)
#pragma unroll
      for (int dc = 0; dc < 2; ++dc)
        kb[half][dc] = nk[half][dc];
  }

  // --- finalize: O /= rowsum, write merged[b*N+q][h*64+d] as bf16 ---
#pragma unroll
  for (int c = 0; c < 4; ++c) {
#pragma unroll
    for (int r = 0; r < 8; ++r) {
      float val = o[c][r] / rsum[r];
      size_t row = (size_t)bb * N_ + qbase + r + 8 * hi;
      merged[row * C_ + hh * D_ + c * 16 + ln] = f2bf(val);
    }
  }
}

// ---------------------------------------------------------------------------
// Host-side orchestration
// ---------------------------------------------------------------------------
extern "C" void kernel_launch(void* const* d_in, const int* in_sizes, int n_in,
                              void* d_out, int out_size, void* d_ws, size_t ws_size,
                              hipStream_t stream) {
  (void)in_sizes; (void)n_in; (void)out_size; (void)ws_size;

  const float* tokens = (const float*)d_in[0];
  const float* pairf  = (const float*)d_in[1];
  const float* Wq     = (const float*)d_in[2];
  const float* bq     = (const float*)d_in[3];
  const float* Wk     = (const float*)d_in[4];
  const float* bk     = (const float*)d_in[5];
  const float* Wv     = (const float*)d_in[6];
  const float* bv     = (const float*)d_in[7];
  const float* Wo     = (const float*)d_in[8];
  const float* bo     = (const float*)d_in[9];
  const float* Wp     = (const float*)d_in[10];

  // Workspace layout (bytes)
  char* w = (char*)d_ws;
  const size_t SZ_TOK  = (size_t)BN_ * C_ * 2;           // 4 MB
  const size_t SZ_W    = (size_t)C_ * C_ * 2;            // 512 KB
  const size_t SZ_QKV  = (size_t)B_ * H_ * N_ * D_ * 2;  // 4 MB
  u16*   tokens_bf = (u16*)(w);                     size_t off = SZ_TOK;
  u16*   WqT = (u16*)(w + off); off += SZ_W;
  u16*   WkT = (u16*)(w + off); off += SZ_W;
  u16*   WvT = (u16*)(w + off); off += SZ_W;
  u16*   WoT = (u16*)(w + off); off += SZ_W;
  u16*   q_bf   = (u16*)(w + off); off += SZ_QKV;
  u16*   k_bf   = (u16*)(w + off); off += SZ_QKV;
  u16*   vT_bf  = (u16*)(w + off); off += SZ_QKV;
  u16*   merged = (u16*)(w + off); off += SZ_TOK;
  float* pbias  = (float*)(w + off);                // 32 MB

  // 1) conversions
  f32_to_bf16_kernel<<<(BN_ * C_ + 255) / 256, 256, 0, stream>>>(tokens, tokens_bf, BN_ * C_);
  transpose_to_bf16_kernel<<<(C_ * C_ + 255) / 256, 256, 0, stream>>>(Wq, WqT, C_, C_);
  transpose_to_bf16_kernel<<<(C_ * C_ + 255) / 256, 256, 0, stream>>>(Wk, WkT, C_, C_);
  transpose_to_bf16_kernel<<<(C_ * C_ + 255) / 256, 256, 0, stream>>>(Wv, WvT, C_, C_);
  transpose_to_bf16_kernel<<<(C_ * C_ + 255) / 256, 256, 0, stream>>>(Wo, WoT, C_, C_);

  // 2) pair bias (streams the 256 MB tensor exactly once)
  pair_bias_kernel<<<(N_ * N_) / 256, 256, 0, stream>>>(pairf, Wp, pbias);

  // 3) Q/K/V projections (WMMA GEMMs)
  dim3 ggrid(BN_ / 128, C_ / 64, 1);
  const float scale = 0.125f;  // 1/sqrt(64), folded into q
  gemm_qkvo_kernel<<<ggrid, 128, 0, stream>>>(tokens_bf, WqT, bq, (void*)q_bf,  0, scale);
  gemm_qkvo_kernel<<<ggrid, 128, 0, stream>>>(tokens_bf, WkT, bk, (void*)k_bf,  1, 1.0f);
  gemm_qkvo_kernel<<<ggrid, 128, 0, stream>>>(tokens_bf, WvT, bv, (void*)vT_bf, 2, 1.0f);

  // 4) flash attention with pair bias
  dim3 agrid(N_ / 64, H_, B_);
  attn_kernel<<<agrid, 128, 0, stream>>>(q_bf, k_bf, vT_bf, pbias, merged);

  // 5) output projection -> d_out (fp32)
  gemm_qkvo_kernel<<<ggrid, 128, 0, stream>>>(merged, WoT, bo, d_out, 3, 1.0f);
}
